// Predictor_64321430225099
// MI455X (gfx1250) — matrix operand order repaired and checked by reading, not compiled, not run on hardware
//
#include <hip/hip_runtime.h>
#include <hip/hip_bf16.h>

// CDNA5 / gfx1250, wave32. WMMA shape 16x16x32 bf16 -> f32 (codegen-confirmed builtin).

typedef __attribute__((ext_vector_type(16))) __bf16 v16bf;
typedef __attribute__((ext_vector_type(8)))  __bf16 v8bf;
typedef __attribute__((ext_vector_type(8)))  float  v8f;

constexpr int B  = 16;
constexpr int F  = 4096;
constexpr int M  = 512;
constexpr int D  = 256;
constexpr int VE = 64;
constexpr int H  = 512;
constexpr int OUT = 8;
constexpr int KC = D + VE;        // 320 = concat width (emb | mean)
constexpr int ROWS = B * M;       // 8192

// ---------------------------------------------------------------------------
// Kernel 0: W_mora [320,512] fp32 -> WbT [512,320] bf16 (transposed so each
// WMMA B-fragment (column n, K contiguous) is two contiguous 16B loads).
// ---------------------------------------------------------------------------
__global__ __launch_bounds__(256) void prep_w_kernel(const float* __restrict__ Wm,
                                                     __bf16* __restrict__ WbT) {
  int t = blockIdx.x * blockDim.x + threadIdx.x;
  if (t >= KC * H) return;
  int n = t / KC;
  int k = t - n * KC;
  WbT[t] = (__bf16)Wm[k * H + n];   // WbT[n][k] = W_mora[k][n]
}

// ---------------------------------------------------------------------------
// Kernel 1: segment mean (sorted indices -> binary search, no atomics) fused
// with embedding gather + concat, emitted as the bf16 WMMA A matrix [8192,320].
// One wave per (b, m): 32 lanes x 8 channels for the mean, 2 for the embedding.
// ---------------------------------------------------------------------------
__global__ __launch_bounds__(256) void seg_mean_kernel(
    const float* __restrict__ feat, const int* __restrict__ midx,
    const int* __restrict__ vow, const float* __restrict__ emb,
    __bf16* __restrict__ A) {
  int wave = threadIdx.x >> 5;
  int lane = threadIdx.x & 31;
  int g = blockIdx.x * 8 + wave;          // 0..8191  (b*M + m)
  int b = g >> 9;                          // /512
  int m = g & 511;

  const int* mi = midx + b * F;            // sorted per utterance
  int lo = 0, hi = F;                      // lower_bound(m)
  while (lo < hi) { int mid = (lo + hi) >> 1; if (mi[mid] < m) lo = mid + 1; else hi = mid; }
  int s0 = lo;
  hi = F;                                  // lower_bound(m+1)
  while (lo < hi) { int mid = (lo + hi) >> 1; if (mi[mid] < m + 1) lo = mid + 1; else hi = mid; }
  int s1 = lo;

  __bf16* arow = A + (size_t)g * KC;

  // columns 0..63: embedding
  int v = vow[g];
  arow[lane]      = (__bf16)emb[v * VE + lane];
  arow[lane + 32] = (__bf16)emb[v * VE + 32 + lane];

  // columns 64..319: mean of frames [s0, s1)
  float s[8];
#pragma unroll
  for (int j = 0; j < 8; ++j) s[j] = 0.f;
  for (int f = s0; f < s1; ++f) {
    const float* fr = feat + ((size_t)b * F + f) * D;
#pragma unroll
    for (int j = 0; j < 8; ++j) s[j] += fr[lane + 32 * j];
  }
  float inv = (s1 > s0) ? 1.f / (float)(s1 - s0) : 0.f;  // empty segment -> 0
#pragma unroll
  for (int j = 0; j < 8; ++j) arow[VE + lane + 32 * j] = (__bf16)(s[j] * inv);
}

// ---------------------------------------------------------------------------
// Kernel 2: mh[8192,512] = A[8192,320](bf16) @ W_mora + b_mora via
// v_wmma_f32_16x16x32_bf16. One 16x16 tile per wave; K-loop of 10 WMMAs.
// Fragment layouts per CDNA5 ISA 7.12.2:
//   A 16x32 bf16: lane l(<16) row l, elems = K[0..7],K[16..23]; lane l+16 row l,
//                 elems = K[8..15],K[24..31]
//   B 32x16 bf16: lane l(<16) col l, K[0..7],K[16..23]; lane l+16 col l, K[8..15],K[24..31]
//   C 16x16 f32 : VGPR r -> row r (lanes 0-15) / row r+8 (lanes 16-31), col = lane%16
// ---------------------------------------------------------------------------
__global__ __launch_bounds__(256) void gemm1_wmma_kernel(
    const __bf16* __restrict__ A, const __bf16* __restrict__ WbT,
    const float* __restrict__ bias, float* __restrict__ mh) {
  int wave = threadIdx.x >> 5;
  int lane = threadIdx.x & 31;
  int t = blockIdx.x * 8 + wave;           // 16384 tiles = 512 x 32
  int tile_m = t >> 5;
  int tile_n = t & 31;
  int row0 = tile_m * 16, col0 = tile_n * 16;
  int lh  = lane & 15;
  int sel = lane >> 4;                     // 0 = lower half lanes, 1 = upper

  v8f acc = {};
  for (int k0 = 0; k0 < KC; k0 += 32) {
    const __bf16* ap = A + (size_t)(row0 + lh) * KC + k0 + sel * 8;
    v8bf a0 = *(const v8bf*)ap;            // K: k0 + sel*8 .. +7
    v8bf a1 = *(const v8bf*)(ap + 16);     // K: k0 + 16 + sel*8 .. +7
    const __bf16* bp = WbT + (size_t)(col0 + lh) * KC + k0 + sel * 8;
    v8bf b0 = *(const v8bf*)bp;            // K: k0 + sel*8 .. +7
    v8bf b1 = *(const v8bf*)(bp + 16);     // K: k0 + 16 + sel*8 .. +7
    v16bf av, bv;
#pragma unroll
    for (int i = 0; i < 8; ++i) {
      av[i] = a0[i]; av[i + 8] = a1[i];
      bv[i] = b0[i]; bv[i + 8] = b1[i];
    }
    acc = __builtin_amdgcn_wmma_f32_16x16x32_bf16(
        /*neg_a=*/false, av, /*neg_b=*/false, bv,
        /*c_mod=*/(short)0, acc, /*reuse_a=*/false, /*reuse_b=*/false);
  }

  float bn = bias[col0 + lh];
#pragma unroll
  for (int r = 0; r < 8; ++r)
    mh[(size_t)(row0 + sel * 8 + r) * H + col0 + lh] = acc[r] + bn;
}

// ---------------------------------------------------------------------------
// Kernel 3: out[8192,8] = mh @ W_post + b_post. Wave per row, lane-strided K,
// xor-shuffle tree reduction (wave32).
// ---------------------------------------------------------------------------
__global__ __launch_bounds__(256) void gemm2_kernel(
    const float* __restrict__ mh, const float* __restrict__ Wp,
    const float* __restrict__ bp, float* __restrict__ out) {
  int wave = threadIdx.x >> 5;
  int lane = threadIdx.x & 31;
  int row = blockIdx.x * 8 + wave;         // 8192 rows
  float acc[OUT];
#pragma unroll
  for (int o = 0; o < OUT; ++o) acc[o] = 0.f;
  const float* mr = mh + (size_t)row * H;
  for (int k = lane; k < H; k += 32) {
    float h = mr[k];
    const float* w = Wp + k * OUT;
#pragma unroll
    for (int o = 0; o < OUT; ++o) acc[o] += h * w[o];
  }
#pragma unroll
  for (int off = 16; off >= 1; off >>= 1) {
#pragma unroll
    for (int o = 0; o < OUT; ++o) acc[o] += __shfl_xor(acc[o], off, 32);
  }
  if (lane < OUT) out[(size_t)row * OUT + lane] = acc[lane] + bp[lane];
}

// ---------------------------------------------------------------------------
extern "C" void kernel_launch(void* const* d_in, const int* in_sizes, int n_in,
                              void* d_out, int out_size, void* d_ws, size_t ws_size,
                              hipStream_t stream) {
  (void)in_sizes; (void)n_in; (void)out_size; (void)ws_size;
  const int*   vowels   = (const int*)d_in[0];
  const float* features = (const float*)d_in[1];
  const int*   midx     = (const int*)d_in[2];
  const float* emb      = (const float*)d_in[3];
  const float* W_mora   = (const float*)d_in[4];
  const float* b_mora   = (const float*)d_in[5];
  // d_in[6], d_in[7] = W_frame, b_frame -> dead code in reference (fh deleted)
  const float* W_post   = (const float*)d_in[8];
  const float* b_post   = (const float*)d_in[9];
  float* out = (float*)d_out;

  char* ws = (char*)d_ws;
  size_t offA   = 0;                                    // A  : 8192*320 bf16 = 5.24 MB
  size_t offWbT = offA + (size_t)ROWS * KC * 2;         // WbT:  512*320 bf16 = 0.33 MB
  size_t offMh  = offWbT + (size_t)H * KC * 2;          // mh : 8192*512 f32  = 16.8 MB
  __bf16* A   = (__bf16*)(ws + offA);
  __bf16* WbT = (__bf16*)(ws + offWbT);
  float*  mh  = (float*)(ws + offMh);

  prep_w_kernel<<<(KC * H + 255) / 256, 256, 0, stream>>>(W_mora, WbT);
  seg_mean_kernel<<<ROWS / 8, 256, 0, stream>>>(features, midx, vowels, emb, A);
  gemm1_wmma_kernel<<<(ROWS / 16) * (H / 16) / 8, 256, 0, stream>>>(A, WbT, b_mora, mh);
  gemm2_kernel<<<ROWS / 8, 256, 0, stream>>>(mh, W_post, b_post, out);
}
